// ModelSimple_27625229648358
// MI455X (gfx1250) — matrix operand order repaired and checked by reading, not compiled, not run on hardware
//
#include <hip/hip_runtime.h>
#include <cstdint>

typedef __attribute__((ext_vector_type(16))) __bf16 v16bf;
typedef __attribute__((ext_vector_type(8)))  float  v8f;
typedef __attribute__((ext_vector_type(4)))  unsigned int u32x4;
typedef __attribute__((ext_vector_type(2)))  unsigned int u32x2;
typedef __attribute__((ext_vector_type(2)))  unsigned short u16x2;
typedef __attribute__((ext_vector_type(4)))  float f32x4;

union FragU { u32x4 q[2]; v16bf v; };

static __device__ __forceinline__ unsigned int f32_to_bf16_bits(float f) {
  union { float f; unsigned u; } c; c.f = f;
  unsigned u = c.u;
  u += 0x7FFFu + ((u >> 16) & 1u);   // round-to-nearest-even
  return u >> 16;
}

// Load one bf16 at uniform base + 32-bit byte offset.
// zext(u32) + uniform base coaxes the GlobalSAddr (saddr+voffset) pattern:
// global_load_*16 v, v_off, s[base:base+1]  -- one 32-bit VALU add per gather.
static __device__ __forceinline__ unsigned short ldg_bf16(
    const unsigned short* __restrict__ base, unsigned byteoff) {
  return *(const unsigned short*)((const char*)base + (size_t)byteoff);
}

// Vectorized f32 -> bf16: float4 in (b128 load), 2x u32 out (b64 store).
__global__ void k_f32_to_bf16x4(const f32x4* __restrict__ in,
                                u32x2* __restrict__ out, int n4) {
  int i = blockIdx.x * blockDim.x + threadIdx.x;
  if (i >= n4) return;
  f32x4 v = in[i];
  u32x2 p;
  p.x = f32_to_bf16_bits(v.x) | (f32_to_bf16_bits(v.y) << 16);
  p.y = f32_to_bf16_bits(v.z) | (f32_to_bf16_bits(v.w) << 16);
  out[i] = p;
}

// Pack weights [COUT, KTOT] f32 -> WMMA-B-fragment layout wt[ks][n(16)][k(32)] bf16.
// Zero padding in N (to 16) and K tail is LOAD-BEARING: the conv kernel gathers
// unmasked (clamped-address) A tiles and relies on zero weights to kill them.
__global__ void k_pack_w(const float* __restrict__ w, unsigned short* __restrict__ wt,
                         int cout, int ktot, int total) {
  int idx = blockIdx.x * blockDim.x + threadIdx.x;
  if (idx >= total) return;
  int ks  = idx >> 9;
  int rem = idx & 511;
  int n   = rem >> 5;
  int k   = rem & 31;
  int kk  = ks * 32 + k;
  float v = (n < cout && kk < ktot) ? w[n * ktot + kk] : 0.0f;
  wt[idx] = (unsigned short)f32_to_bf16_bits(v);
}

// tap index kk (ic-major, then kh,kw,kd,kt over 7^4) -> input element offset
template <int IS, int ISP4>
static __device__ __forceinline__ int tap_off(int kk) {
  int ic = kk / 2401;
  int t  = kk - ic * 2401;
  int kh = t / 343;  t -= kh * 343;
  int kw = t / 49;   t -= kw * 49;
  int kd = t / 7;    int kt = t - kd * 7;
  return ic * ISP4 + ((kh * IS + kw) * IS + kd) * IS + kt;
}

// 4D conv (valid, stride 1) as im2col GEMM on bf16 WMMA, f32 accumulate.
// One wave per 32-position x 16-channel output tile (2 WMMA M-tiles sharing the
// B fragment and the per-kstep tap decomposition); double-buffered LDS A tile.
template <int IS, int OS, int CIN, int COUT, int KSTEPS, int KTOT, bool OUT_BF16>
__global__ void __launch_bounds__(32) k_conv4d_wmma(
    const unsigned short* __restrict__ xin,  // bf16 [B][CIN][IS^4]
    const unsigned short* __restrict__ wt,   // bf16 [KSTEPS][16][32]
    const float* __restrict__ bias,          // [COUT]
    void* __restrict__ outp)                 // [B][COUT][OS^4]
{
  constexpr int ISP4 = IS * IS * IS * IS;
  constexpr int OSP4 = OS * OS * OS * OS;

  const int lane    = threadIdx.x & 31;
  const int half    = lane >> 4;             // 0: lanes 0-15, 1: lanes 16-31
  const int b       = blockIdx.y;
  const int posbase = blockIdx.x * 32;       // 2 M-tiles of 16 positions

  __shared__ __align__(16) unsigned short tileA[2][32 * 32];
  __shared__ int basePs[32];

  {
    int p = posbase + lane;
    if (p >= OSP4) p = OSP4 - 1;             // phantom-tile clamp (never stored)
    int ot = p % OS; p /= OS;
    int od = p % OS; p /= OS;
    int ow = p % OS; int oh = p / OS;
    basePs[lane] = ((oh * IS + ow) * IS + od) * IS + ot;
  }
  __syncthreads();

  // Gather role: this lane fills k-pair kp (K columns 2kp, 2kp+1) of rows
  // rb..rb+7 of tile0 and 16+rb..16+rb+7 of tile1. Byte offsets, u32.
  const int kp = lane & 15;
  const int rb = half * 8;
  unsigned bplB0[8], bplB1[8];
  #pragma unroll
  for (int r = 0; r < 8; ++r) {
    bplB0[r] = (unsigned)basePs[rb + r] * 2u;
    bplB1[r] = (unsigned)basePs[16 + rb + r] * 2u;
  }

  const unsigned short* img = xin + (size_t)b * (CIN * ISP4);

  v8f acc0 = {0.f, 0.f, 0.f, 0.f, 0.f, 0.f, 0.f, 0.f};
  v8f acc1 = {0.f, 0.f, 0.f, 0.f, 0.f, 0.f, 0.f, 0.f};

  for (int ks = 0; ks < KSTEPS; ++ks) {
    unsigned short* tb = &tileA[ks & 1][0];

    // One tap decomposition pair per k-step, amortized over 2 WMMAs.
    // Branchless: clamp tap index; K-tail garbage annihilated by zero weights.
    int kk0 = ks * 32 + kp * 2;
    int c0 = (kk0     < KTOT) ? kk0     : (KTOT - 1);
    int c1 = (kk0 + 1 < KTOT) ? kk0 + 1 : (KTOT - 1);
    const unsigned t0B = (unsigned)tap_off<IS, ISP4>(c0) * 2u;
    const unsigned t1B = (unsigned)tap_off<IS, ISP4>(c1) * 2u;

    #pragma unroll
    for (int r = 0; r < 8; ++r) {
      u16x2 v0, v1;                           // d16 / d16_hi load pairing
      v0.x = ldg_bf16(img, t0B + bplB0[r]);
      v0.y = ldg_bf16(img, t1B + bplB0[r]);
      v1.x = ldg_bf16(img, t0B + bplB1[r]);
      v1.y = ldg_bf16(img, t1B + bplB1[r]);
      *(u16x2*)&tb[(rb + r) * 32 + kp * 2]      = v0;  // ds_store_b32 (2addr-merged)
      *(u16x2*)&tb[(16 + rb + r) * 32 + kp * 2] = v1;
    }
    __syncthreads();   // single-wave WG: lowers to s_wait_dscnt; ping-pong buffer
                       // makes one wait point per iteration sufficient

    // A fragments (CDNA5 16-bit A 16x32): lane M=lane&15,
    // K chunks [kb..kb+7] and [kb+16..kb+23], kb = 8*half. Two ds_load_b128 each.
    FragU a0, a1, bf;
    const int m  = lane & 15;
    const int kb = half * 8;
    a0.q[0] = *(const u32x4*)&tb[m * 32 + kb];
    a0.q[1] = *(const u32x4*)&tb[m * 32 + kb + 16];
    a1.q[0] = *(const u32x4*)&tb[(16 + m) * 32 + kb];
    a1.q[1] = *(const u32x4*)&tb[(16 + m) * 32 + kb + 16];

    // B fragment (16-bit B 32x16): lane N=lane&15, K chunks [kb..kb+7] and [kb+16..kb+23], from packed
    // global weights (L2-resident, shared by all blocks AND both WMMAs below).
    const unsigned short* wrow = wt + (size_t)ks * 512 + (size_t)m * 32 + half * 8;
    bf.q[0] = *(const u32x4*)(wrow);
    bf.q[1] = *(const u32x4*)(wrow + 16);

    acc0 = __builtin_amdgcn_wmma_f32_16x16x32_bf16(false, a0.v, false, bf.v,
                                                   (short)0, acc0, false, false);
    acc1 = __builtin_amdgcn_wmma_f32_16x16x32_bf16(false, a1.v, false, bf.v,
                                                   (short)0, acc1, false, false);
  }

  // C tiles: lane N = lane&15; VGPR r -> M = r + 8*half.
  const int n = lane & 15;
  if (n < COUT) {
    const float bn = bias[n];
    #pragma unroll
    for (int r = 0; r < 8; ++r) {
      int   m = half * 8 + r;
      float v = acc0[r] + bn;
      v = v > 0.f ? v : 0.f;                  // ReLU
      size_t o = ((size_t)b * COUT + n) * OSP4 + (size_t)(posbase + m);
      if (OUT_BF16) ((unsigned short*)outp)[o] = (unsigned short)f32_to_bf16_bits(v);
      else          ((float*)outp)[o] = v;
    }
    if (posbase + 16 < OSP4) {                // tile1 may be phantom (conv2 tail)
      #pragma unroll
      for (int r = 0; r < 8; ++r) {
        int   m = 16 + half * 8 + r;
        float v = acc1[r] + bn;
        v = v > 0.f ? v : 0.f;
        size_t o = ((size_t)b * COUT + n) * OSP4 + (size_t)(posbase + m);
        if (OUT_BF16) ((unsigned short*)outp)[o] = (unsigned short)f32_to_bf16_bits(v);
        else          ((float*)outp)[o] = v;
      }
    }
  }
}

__global__ void __launch_bounds__(256) k_linear_sigmoid(
    const float* __restrict__ h2, const float* __restrict__ wl,
    const float* __restrict__ bl, float* __restrict__ out, int feat) {
  __shared__ float red[256];
  const int b = blockIdx.x;
  float s = 0.f;
  for (int i = threadIdx.x; i < feat; i += 256)
    s += h2[(size_t)b * feat + i] * wl[i];
  red[threadIdx.x] = s;
  __syncthreads();
  for (int w = 128; w > 0; w >>= 1) {
    if ((int)threadIdx.x < w) red[threadIdx.x] += red[threadIdx.x + w];
    __syncthreads();
  }
  if (threadIdx.x == 0) {
    float z = red[0] + bl[0];
    out[b] = 1.f / (1.f + __expf(-z));
  }
}

extern "C" void kernel_launch(void* const* d_in, const int* in_sizes, int n_in,
                              void* d_out, int out_size, void* d_ws, size_t ws_size,
                              hipStream_t stream) {
  const float* x  = (const float*)d_in[0];   // [128,1,18,18,18,18]
  const float* w1 = (const float*)d_in[1];   // [5,1,7,7,7,7]
  const float* b1 = (const float*)d_in[2];   // [5]
  const float* w2 = (const float*)d_in[3];   // [10,5,7,7,7,7]
  const float* b2 = (const float*)d_in[4];   // [10]
  const float* wl = (const float*)d_in[5];   // [1,12960]
  const float* bl = (const float*)d_in[6];   // [1]
  float* out = (float*)d_out;                // [128,1] f32
  (void)in_sizes; (void)n_in; (void)out_size; (void)ws_size;

  const int B    = 128;
  const int NX   = B * 18 * 18 * 18 * 18;    // 13,436,928 input elems (mult of 4)
  const int OSP1 = 12 * 12 * 12 * 12;        // 20736
  const int OSP2 = 6 * 6 * 6 * 6;            // 1296
  const int KS1  = 76,  KT1 = 2401;          // ceil(2401/32)
  const int KS2  = 376, KT2 = 12005;         // ceil(12005/32)
  const int FEAT = 10 * OSP2;                // 12960

  char* ws = (char*)d_ws;
  size_t off = 0;
  auto carve = [&](size_t bytes) {
    char* p = ws + off;
    off += (bytes + 255) & ~(size_t)255;
    return p;
  };
  unsigned short* xb  = (unsigned short*)carve((size_t)NX * 2);            // x as bf16
  unsigned short* w1t = (unsigned short*)carve((size_t)KS1 * 512 * 2);     // packed W1
  unsigned short* w2t = (unsigned short*)carve((size_t)KS2 * 512 * 2);     // packed W2
  unsigned short* h1  = (unsigned short*)carve((size_t)B * 5 * OSP1 * 2);  // bf16 act
  float*          h2  = (float*)carve((size_t)B * 10 * OSP2 * 4);          // f32 act

  const int N4 = NX / 4;
  k_f32_to_bf16x4<<<(N4 + 255) / 256, 256, 0, stream>>>((const f32x4*)x, (u32x2*)xb, N4);
  k_pack_w<<<(KS1 * 512 + 255) / 256, 256, 0, stream>>>(w1, w1t, 5,  KT1, KS1 * 512);
  k_pack_w<<<(KS2 * 512 + 255) / 256, 256, 0, stream>>>(w2, w2t, 10, KT2, KS2 * 512);

  // grids: ceil(OSP/32) position-pair tiles per image
  k_conv4d_wmma<18, 12, 1, 5,  KS1, KT1, true>
      <<<dim3((OSP1 + 31) / 32, B), 32, 0, stream>>>(xb, w1t, b1, (void*)h1);
  k_conv4d_wmma<12, 6,  5, 10, KS2, KT2, false>
      <<<dim3((OSP2 + 31) / 32, B), 32, 0, stream>>>(h1, w2t, b2, (void*)h2);

  k_linear_sigmoid<<<B, 256, 0, stream>>>(h2, wl, bl, out, FEAT);
}
